// TARNet_23313082482710
// MI455X (gfx1250) — matrix operand order repaired and checked by reading, not compiled, not run on hardware
//
#include <hip/hip_runtime.h>

// ---------------- problem constants ----------------
constexpr int Bn  = 16384;   // batch
constexpr int KIN = 512;     // input features
constexpr int H1  = 512;
constexpr int H2  = 256;
constexpr int NO  = 4;       // outcomes
constexpr int NT  = 4;       // treatments

constexpr int BM   = 64;     // batch rows per workgroup
constexpr int SLDS = 520;    // LDS row stride (elements), pad 8 -> 16 distinct banks

typedef __attribute__((ext_vector_type(16))) __bf16 v16bf;
typedef __attribute__((ext_vector_type(8)))  __bf16 v8bf;
typedef __attribute__((ext_vector_type(8)))  float  v8f;

__device__ __forceinline__ unsigned short f2bf(float f) {
  unsigned int u = __float_as_uint(f);
  unsigned int r = u + 0x7FFFu + ((u >> 16) & 1u);   // round-to-nearest-even
  return (unsigned short)(r >> 16);
}

// ---------------- prep kernels ----------------

// feature fp32 -> bf16, 4 elements / thread
__global__ void k_feat(const float* __restrict__ f, unsigned short* __restrict__ fb) {
  int i = blockIdx.x * blockDim.x + threadIdx.x;          // over Bn*KIN/4
  float4 v = ((const float4*)f)[i];
  ushort4 r;
  r.x = f2bf(v.x); r.y = f2bf(v.y); r.z = f2bf(v.z); r.w = f2bf(v.w);
  ((ushort4*)fb)[i] = r;
}

// W1 [O,KIN,H1] fp32 -> packed bf16 WMMA B-fragments.
// Per o: tiles tIdx = nt*16 + kt (nt over H1/16, kt over KIN/32), each tile
// = 32 lanes * 16 bf16: lane l holds N = nt*16 + l%16, K = kt*32 + (l>=16)*8 + (j&7) + (j>=8)*16.
__global__ void k_w1p(const float* __restrict__ W1, unsigned short* __restrict__ w1p) {
  int i = blockIdx.x * blockDim.x + threadIdx.x;          // over NO*512*512 = 1,048,576
  int j    = i & 15;
  int l    = (i >> 4) & 31;
  int tIdx = (i >> 9) & 511;
  int o    = i >> 18;
  int kt = tIdx & 15, nt = tIdx >> 4;
  int n = nt * 16 + (l & 15);
  int k = kt * 32 + ((l >> 4) << 3) + (j & 7) + ((j >> 3) << 4);
  w1p[i] = f2bf(W1[(o * KIN + k) * H1 + n]);
}

// W2 [O,T,H1,H2] fp32 -> packed bf16 WMMA B-fragments (per ot: 256 tiles).
__global__ void k_w2p(const float* __restrict__ W2, unsigned short* __restrict__ w2p) {
  int i = blockIdx.x * blockDim.x + threadIdx.x;          // over 16*256*512 = 2,097,152
  int j    = i & 15;
  int l    = (i >> 4) & 31;
  int tIdx = (i >> 9) & 255;
  int ot   = i >> 17;
  int kt = tIdx & 15, nt = tIdx >> 4;
  int n = nt * 16 + (l & 15);
  int k = kt * 32 + ((l >> 4) << 3) + (j & 7) + ((j >> 3) << 4);
  w2p[i] = f2bf(W2[(ot * H1 + k) * H2 + n]);
}

// fold BN: y = x*s + t with s = g*rsqrt(v+eps), t = be - m*s
__global__ void k_fold(const float* g1, const float* be1, const float* m1, const float* v1,
                       const float* g2, const float* be2, const float* m2, const float* v2,
                       float* s1f, float* t1f, float* s2f, float* t2f) {
  int i = blockIdx.x * blockDim.x + threadIdx.x;
  if (i < NO * H1)      { float s = g1[i] * rsqrtf(v1[i] + 1e-5f); s1f[i] = s; t1f[i] = be1[i] - m1[i] * s; }
  if (i < NO * NT * H2) { float s = g2[i] * rsqrtf(v2[i] + 1e-5f); s2f[i] = s; t2f[i] = be2[i] - m2[i] * s; }
}

// load one A-fragment (16x32 bf16) for M-subtile m at K-offset ka from LDS tile
__device__ __forceinline__ v16bf load_afrag(const unsigned short* s, int m, int laneN, int ka) {
  const v8bf lo = *(const v8bf*)(s + (m * 16 + laneN) * SLDS + ka);
  const v8bf hh = *(const v8bf*)(s + (m * 16 + laneN) * SLDS + ka + 16);
  return __builtin_shufflevector(lo, hh, 0,1,2,3,4,5,6,7,8,9,10,11,12,13,14,15);
}

// ---------------- fused TARNet forward ----------------
__global__ __launch_bounds__(256)
void tarnet_fused(const unsigned short* __restrict__ featB,
                  const unsigned short* __restrict__ w1p,
                  const unsigned short* __restrict__ w2p,
                  const float* __restrict__ b1,
                  const float* __restrict__ s1f, const float* __restrict__ t1f,
                  const float* __restrict__ b2,
                  const float* __restrict__ s2f, const float* __restrict__ t2f,
                  const float* __restrict__ W3, const float* __restrict__ b3,
                  const int* __restrict__ treat, float* __restrict__ out)
{
  extern __shared__ __align__(16) unsigned short smem[];
  unsigned short* sF    = smem;                 // BM x SLDS bf16 (feature tile)
  unsigned short* sH    = smem + BM * SLDS;     // BM x SLDS bf16 (hidden tile)
  float*          sPred = (float*)(sH + BM * SLDS);  // BM x NT

  const int o     = blockIdx.y;
  const int b0    = blockIdx.x * BM;
  const int tid   = threadIdx.x;
  const int lane  = tid & 31;
  const int wid   = tid >> 5;
  const int laneN = lane & 15;
  const int hi    = lane >> 4;        // 0 / 1 half-wave

  // ---- stage 0: async-DMA feature tile into LDS (16B per lane per op) ----
  for (int c = tid; c < BM * (KIN / 8); c += 256) {
    const int row = c >> 6;           // KIN/8 == 64 chunks per row
    const int col = (c & 63) << 3;
    const unsigned ldsoff = (unsigned)(size_t)(sF + row * SLDS + col);
    const unsigned long long ga =
        (unsigned long long)(size_t)(featB + (b0 + row) * KIN + col);
    asm volatile("global_load_async_to_lds_b128 %0, %1, off"
                 :: "v"(ldsoff), "v"(ga) : "memory");
  }
  if (tid < BM * NT) sPred[tid] = b3[o * NT + (tid & (NT - 1))];
  asm volatile("s_wait_asynccnt 0x0" ::: "memory");
  __syncthreads();

  // ---- stage 1: h = BN(relu(x @ W1 + b1)) -> LDS bf16 ----
  // Each wave owns 64 cols of H1 (4 N-subtiles), all 4 M-subtiles register-
  // blocked: A-fragments loaded ONCE per K-step, 16 WMMAs per 8 ds_loads.
  {
    v8f acc[16] = {};                      // [ns*4 + m]
    const int nt0 = wid * 4;               // first N-tile of this wave
    const unsigned short* bp = w1p + (o * 512 + nt0 * 16) * 512 + lane * 16;
    #pragma unroll 4
    for (int kt = 0; kt < 16; ++kt) {
      const int ka = kt * 32 + hi * 8;
      v16bf a[4];
      #pragma unroll
      for (int m = 0; m < 4; ++m) a[m] = load_afrag(sF, m, laneN, ka);
      #pragma unroll
      for (int ns = 0; ns < 4; ++ns) {
        const v16bf bf = *(const v16bf*)(bp + (ns * 16 + kt) * 512);
        #pragma unroll
        for (int m = 0; m < 4; ++m)
          acc[ns * 4 + m] = __builtin_amdgcn_wmma_f32_16x16x32_bf16(
              false, a[m], false, bf, (short)0, acc[ns * 4 + m], false, false);
      }
    }
    #pragma unroll
    for (int ns = 0; ns < 4; ++ns) {
      const int n  = (nt0 + ns) * 16 + laneN;
      const float bb = b1[o * H1 + n];
      const float ss = s1f[o * H1 + n];
      const float tt = t1f[o * H1 + n];
      #pragma unroll
      for (int m = 0; m < 4; ++m)
        #pragma unroll
        for (int r = 0; r < 8; ++r) {
          const int row = m * 16 + hi * 8 + r;
          const float hv = fmaxf(acc[ns * 4 + m][r] + bb, 0.f) * ss + tt;
          sH[row * SLDS + n] = f2bf(hv);
        }
    }
  }
  __syncthreads();

  // ---- stage 2: all T heads: z = BN(relu(h @ W2 + b2)); pred += z . W3 ----
  // Concatenated N = T*H2 = 1024; wave owns 128 cols (one treatment per wave
  // pair), processed as 2 register-blocked passes of 4 N-subtiles.
  {
    float partial[32];
    #pragma unroll
    for (int i = 0; i < 32; ++i) partial[i] = 0.f;
    const int t = wid >> 1;                // constant per wave
    for (int pass = 0; pass < 2; ++pass) {
      const int ngBase = wid * 128 + pass * 64;
      const int ntBase = (ngBase & 255) >> 4;
      const unsigned short* bp =
          w2p + ((o * NT + t) * 256 + ntBase * 16) * 512 + lane * 16;
      v8f acc[16] = {};
      #pragma unroll 4
      for (int kt = 0; kt < 16; ++kt) {
        const int ka = kt * 32 + hi * 8;
        v16bf a[4];
        #pragma unroll
        for (int m = 0; m < 4; ++m) a[m] = load_afrag(sH, m, laneN, ka);
        #pragma unroll
        for (int ns = 0; ns < 4; ++ns) {
          const v16bf bf = *(const v16bf*)(bp + (ns * 16 + kt) * 512);
          #pragma unroll
          for (int m = 0; m < 4; ++m)
            acc[ns * 4 + m] = __builtin_amdgcn_wmma_f32_16x16x32_bf16(
                false, a[m], false, bf, (short)0, acc[ns * 4 + m], false, false);
        }
      }
      #pragma unroll
      for (int ns = 0; ns < 4; ++ns) {
        const int n  = (ngBase & 255) + ns * 16 + laneN;
        const int ci = (o * NT + t) * H2 + n;
        const float bb = b2[ci], ss = s2f[ci], tt = t2f[ci], w3v = W3[ci];
        #pragma unroll
        for (int m = 0; m < 4; ++m)
          #pragma unroll
          for (int r = 0; r < 8; ++r) {
            const float z = fmaxf(acc[ns * 4 + m][r] + bb, 0.f) * ss + tt;
            partial[m * 8 + r] += z * w3v;
          }
      }
    }
    #pragma unroll
    for (int m = 0; m < 4; ++m)
      #pragma unroll
      for (int r = 0; r < 8; ++r) {
        const int row = m * 16 + hi * 8 + r;
        atomicAdd(&sPred[row * NT + t], partial[m * 8 + r]);
      }
  }
  __syncthreads();

  // ---- route by treatment ----
  if (tid < BM) {
    const int b  = b0 + tid;
    const int tr = treat[b];
    out[b * NO + o] = sPred[tid * NT + tr];
  }
}

// ---------------- launcher ----------------
extern "C" void kernel_launch(void* const* d_in, const int* in_sizes, int n_in,
                              void* d_out, int out_size, void* d_ws, size_t ws_size,
                              hipStream_t stream) {
  const float* feature = (const float*)d_in[0];
  const int*   treat   = (const int*)  d_in[1];
  const float* W1 = (const float*)d_in[2];
  const float* b1 = (const float*)d_in[3];
  const float* g1 = (const float*)d_in[4];
  const float* be1= (const float*)d_in[5];
  const float* m1 = (const float*)d_in[6];
  const float* v1 = (const float*)d_in[7];
  const float* W2 = (const float*)d_in[8];
  const float* b2 = (const float*)d_in[9];
  const float* g2 = (const float*)d_in[10];
  const float* be2= (const float*)d_in[11];
  const float* m2 = (const float*)d_in[12];
  const float* v2 = (const float*)d_in[13];
  const float* W3 = (const float*)d_in[14];
  const float* b3 = (const float*)d_in[15];

  char* ws = (char*)d_ws;
  unsigned short* featB = (unsigned short*)(ws);                       // 16 MB
  unsigned short* w1p   = (unsigned short*)(ws + 16777216);            //  2 MB
  unsigned short* w2p   = (unsigned short*)(ws + 18874368);            //  4 MB
  float* s1f = (float*)(ws + 23068672);
  float* t1f = s1f + NO * H1;
  float* s2f = t1f + NO * H1;
  float* t2f = s2f + NO * NT * H2;

  k_feat<<<(Bn * KIN / 4) / 256, 256, 0, stream>>>(feature, featB);
  k_w1p <<<(NO * H1 * KIN) / 256, 256, 0, stream>>>(W1, w1p);
  k_w2p <<<(NO * NT * H2 * H1) / 256, 256, 0, stream>>>(W2, w2p);
  k_fold<<<16, 256, 0, stream>>>(g1, be1, m1, v1, g2, be2, m2, v2, s1f, t1f, s2f, t2f);

  const size_t shmem = (size_t)BM * SLDS * 2 * 2 + (size_t)BM * NT * 4;  // 134,144 B
  tarnet_fused<<<dim3(Bn / BM, NO), 256, shmem, stream>>>(
      featB, w1p, w2p, b1, s1f, t1f, b2, s2f, t2f, W3, b3, treat, (float*)d_out);
}
